// AdaptiveHighPassFilter_14886356648173
// MI455X (gfx1250) — matrix operand order repaired and checked by reading, not compiled, not run on hardware
//
#include <hip/hip_runtime.h>

typedef __attribute__((ext_vector_type(16))) _Float16 v16h;
typedef __attribute__((ext_vector_type(8)))  _Float16 v8h;
typedef __attribute__((ext_vector_type(8)))  float    v8f;
typedef __attribute__((ext_vector_type(4)))  float    v4f;

#define TILE   16
#define PATCH  18            // TILE + 2 halo
#define C_IN   64
#define HGT    128
#define WID    128
#define F32_STRIDE 68        // floats per patch pixel (padded: conflict-free v4f reads)
#define F16_STRIDE 72        // halves per patch pixel (padded: conflict-free b128 reads)

// LDS layout (bytes)
#define OFF_F32 0
#define SZ_F32  (PATCH*PATCH*F32_STRIDE*4)     // 88128
#define OFF_F16 (OFF_F32 + SZ_F32)
#define SZ_F16  (PATCH*PATCH*F16_STRIDE*2)     // 46656
#define OFF_B   (OFF_F16 + SZ_F16)             // 134784
#define SZ_B    (18*1024)                      // 9 taps x 2 chunks x 1KB B-fragments
#define OFF_M   (OFF_B + SZ_B)                 // 153216
#define SZ_M    (256*17*4)                     // per-pixel logits/weights, stride 17
#define SMEM_BYTES (OFF_M + SZ_M)              // 170624 (< 320KB WGP limit)

static_assert(SMEM_BYTES <= 320*1024, "LDS budget");
static_assert((OFF_F16 % 16) == 0 && (OFF_B % 16) == 0 && (OFF_M % 16) == 0, "align");

__device__ __forceinline__ int reflect_idx(int i, int n) {
  if (i < 0)  i = -i;
  if (i >= n) i = 2*n - 2 - i;
  return i;
}

__global__ __launch_bounds__(256)
void ahpf_fused_kernel(const float* __restrict__ x, const float* __restrict__ W,
                       const float* __restrict__ bias, float* __restrict__ out) {
  extern __shared__ char smem[];
  float*    f32p = (float*)(smem + OFF_F32);
  _Float16* f16p = (_Float16*)(smem + OFF_F16);
  _Float16* bfr  = (_Float16*)(smem + OFF_B);
  float*    mlds = (float*)(smem + OFF_M);

  const int tid  = threadIdx.x;
  const int wave = tid >> 5;
  const int lane = tid & 31;
  const int b    = blockIdx.y;           // batch
  const int ty   = blockIdx.x >> 3;
  const int tx   = blockIdx.x & 7;
  const int y0   = ty * TILE;
  const int x0   = tx * TILE;

  // ---- Stage 1: load 18x18x64 reflect-padded patch -> LDS (f32 + f16) ----
  const float* xb = x + (size_t)b * C_IN * (HGT*WID);
  for (int idx = tid; idx < C_IN*PATCH*PATCH; idx += 256) {   // 81 iters, exact
    int xx = idx % PATCH;
    int t  = idx / PATCH;
    int yy = t % PATCH;
    int c  = t / PATCH;
    int gy = reflect_idx(y0 - 1 + yy, HGT);
    int gx = reflect_idx(x0 - 1 + xx, WID);
    float v = xb[c*(HGT*WID) + gy*WID + gx];
    int pix = yy*PATCH + xx;
    f32p[pix*F32_STRIDE + c] = v;
    f16p[pix*F16_STRIDE + c] = (_Float16)v;
  }

  // ---- Stage 2: build B fragments (W as f16, WMMA B layout) --------------
  {
    int n  = lane & 15;                  // output channel (N), 9 valid
    int hi = lane >> 4;                  // half-wave -> K half
    for (int f = wave; f < 18; f += 8) {
      int tap = f >> 1, chunk = f & 1;
      int ti = tap / 3, tj = tap % 3;
      v16h bv;
      #pragma unroll
      for (int h = 0; h < 16; ++h) {
        int c = chunk*32 + hi*16 + h;    // K index == input channel
        float wv = (n < 9) ? W[((n*C_IN + c)*3 + ti)*3 + tj] : 0.0f;
        bv[h] = (_Float16)wv;
      }
      *(v16h*)(bfr + f*512 + lane*16) = bv;
    }
  }
  __syncthreads();

  // ---- Stage 3: 3x3x64 -> 9 conv as WMMA f16 (A = shifted x patches) -----
  {
    int m  = lane & 15;                  // A row M == px; also N col for C store
    int hi = lane >> 4;
    #pragma unroll
    for (int gi = 0; gi < 2; ++gi) {
      int g = wave*2 + gi;               // tile row (pixel group) for this wave
      v8f acc = {};
      #pragma unroll
      for (int tap = 0; tap < 9; ++tap) {
        int ti = tap / 3, tj = tap % 3;
        int pix = (g + ti)*PATCH + (m + tj);
        #pragma unroll
        for (int chunk = 0; chunk < 2; ++chunk) {
          // A layout matches B: halves0-15 K=base..base+15 (base = hi*16)
          const _Float16* ap = f16p + pix*F16_STRIDE + chunk*32 + hi*16;
          v8h a0 = *(const v8h*)ap;
          v8h a1 = *(const v8h*)(ap + 8);
          v16h A;
          #pragma unroll
          for (int h = 0; h < 8; ++h) { A[h] = a0[h]; A[8+h] = a1[h]; }
          v16h B = *(const v16h*)(bfr + (tap*2 + chunk)*512 + lane*16);
          acc = __builtin_amdgcn_wmma_f32_16x16x32_f16(
              false, A, false, B, (short)0, acc, false, false);
        }
      }
      // C layout: VGPR r -> M=r(+8 for hi half), N=lane%16
      #pragma unroll
      for (int r = 0; r < 8; ++r) {
        int p = g*16 + r + hi*8;         // pixel index within tile
        mlds[p*17 + m] = acc[r];
      }
    }
  }
  __syncthreads();

  // ---- Stage 4: softmax + hamming + renormalize (1 pixel / thread) -------
  {
    const float ham[9] = {0.0064f, 0.08f, 0.0064f,
                          0.08f,   1.0f,  0.08f,
                          0.0064f, 0.08f, 0.0064f};
    float l[9];
    #pragma unroll
    for (int i = 0; i < 9; ++i) l[i] = mlds[tid*17 + i] + bias[i];
    float mx = l[0];
    #pragma unroll
    for (int i = 1; i < 9; ++i) mx = fmaxf(mx, l[i]);
    float s = 0.f;
    #pragma unroll
    for (int i = 0; i < 9; ++i) { l[i] = __expf(l[i] - mx) * ham[i]; s += l[i]; }
    float inv = 1.0f / s;                // softmax denom cancels; one renorm
    #pragma unroll
    for (int i = 0; i < 9; ++i) mlds[tid*17 + i] = l[i] * inv;
  }
  __syncthreads();

  // ---- Stage 5: CARAFE weighted 3x3 sum + highpass residual --------------
  {
    int py = tid >> 4, px = tid & 15;
    float wgt[9];
    #pragma unroll
    for (int i = 0; i < 9; ++i) wgt[i] = mlds[tid*17 + i];
    int gy = y0 + py, gx = x0 + px;
    float* ob = out + (size_t)b * C_IN * (HGT*WID) + gy*WID + gx;
    #pragma unroll 2
    for (int c = 0; c < C_IN; c += 4) {
      v4f acc = {0.f, 0.f, 0.f, 0.f};
      v4f ctr = {0.f, 0.f, 0.f, 0.f};
      #pragma unroll
      for (int tap = 0; tap < 9; ++tap) {
        int ti = tap / 3, tj = tap % 3;
        v4f v = *(const v4f*)(f32p + ((py+ti)*PATCH + (px+tj))*F32_STRIDE + c);
        acc += wgt[tap] * v;
        if (tap == 4) ctr = v;
      }
      v4f o = 2.0f*ctr - acc;            // 2x - lowpass
      #pragma unroll
      for (int i = 0; i < 4; ++i) ob[(c+i)*(HGT*WID)] = o[i];
    }
  }
}

extern "C" void kernel_launch(void* const* d_in, const int* in_sizes, int n_in,
                              void* d_out, int out_size, void* d_ws, size_t ws_size,
                              hipStream_t stream) {
  (void)in_sizes; (void)n_in; (void)d_ws; (void)ws_size; (void)out_size;
  const float* x  = (const float*)d_in[0];
  const float* W  = (const float*)d_in[1];
  const float* bi = (const float*)d_in[2];
  float* out = (float*)d_out;
  dim3 grid(64, 8, 1);                   // 8x8 tiles of 16x16, 8 batches
  ahpf_fused_kernel<<<grid, 256, SMEM_BYTES, stream>>>(x, W, bi, out);
}
